// SplitGPT2_2104533975473
// MI455X (gfx1250) — matrix operand order repaired and checked by reading, not compiled, not run on hardware
//
#include <hip/hip_runtime.h>
#include <hip/hip_bf16.h>
#include <math.h>

// ---------- model dims ----------
#define B_   4
#define L_   1024
#define D_   1024
#define H_   16
#define HD_  64
#define NL_  3
#define R_   4
#define MTOK (B_*L_)          // 4096 tokens

// ---------- types for WMMA ----------
typedef __bf16 bf16;
typedef __attribute__((ext_vector_type(16))) __bf16 v16bf;
typedef __attribute__((ext_vector_type(8)))  float  v8f;
typedef __attribute__((ext_vector_type(4)))  unsigned int u32x4;

union FragBF { v16bf v; u32x4 u[2]; };

__device__ inline bf16 f2bf(float f) {
  unsigned u = __float_as_uint(f);
  unsigned short s = (unsigned short)((u + 0x7FFFu + ((u >> 16) & 1u)) >> 16);
  return __builtin_bit_cast(bf16, s);
}
__device__ inline float bf2f(bf16 h) {
  unsigned short s = __builtin_bit_cast(unsigned short, h);
  return __uint_as_float(((unsigned)s) << 16);
}
__device__ inline float gelu_f(float x) {
  const float c = 0.7978845608028654f;   // sqrt(2/pi)
  float t = tanhf(c * (x + 0.044715f * x * x * x));
  return 0.5f * x * (1.0f + t);
}

// ---------- CDNA5 async global->LDS (DMA, tracked by ASYNCcnt) ----------
// LDS operand VGPR holds the LDS byte offset (low 32 bits of generic addr).
__device__ inline void async_ld_b128(const bf16* g, const bf16* l) {
  unsigned lds = (unsigned)(size_t)l;
  unsigned long long ga = (unsigned long long)g;
  asm volatile("global_load_async_to_lds_b128 %0, %1, off"
               :: "v"(lds), "v"(ga) : "memory");
}

// ---------- embedding: x = wte[ids] + wpe[pos] ----------
__global__ void k_embed(const int* __restrict__ ids, const float* __restrict__ wte,
                        const float* __restrict__ wpe, float* __restrict__ x) {
  int t  = blockIdx.x;            // token 0..MTOK-1
  int l  = t & (L_ - 1);
  int id = ids[t];
  const float* a = wte + (size_t)id * D_;
  const float* p = wpe + (size_t)l  * D_;
  float* o = x + (size_t)t * D_;
  for (int i = threadIdx.x; i < D_; i += 256) o[i] = a[i] + p[i];
}

// ---------- layernorm -> bf16 ----------
__global__ void k_ln(const float* __restrict__ x, const float* __restrict__ w,
                     const float* __restrict__ b, bf16* __restrict__ out) {
  int row = blockIdx.x;
  const float* xr = x + (size_t)row * D_;
  float s = 0.f, s2 = 0.f;
  for (int i = threadIdx.x; i < D_; i += 256) { float v = xr[i]; s += v; s2 += v * v; }
  __shared__ float r1[256], r2[256];
  r1[threadIdx.x] = s; r2[threadIdx.x] = s2;
  __syncthreads();
  for (int off = 128; off > 0; off >>= 1) {
    if (threadIdx.x < off) { r1[threadIdx.x] += r1[threadIdx.x + off];
                             r2[threadIdx.x] += r2[threadIdx.x + off]; }
    __syncthreads();
  }
  float mean = r1[0] * (1.0f / D_);
  float var  = r2[0] * (1.0f / D_) - mean * mean;
  float inv  = rsqrtf(var + 1e-5f);
  bf16* orow = out + (size_t)row * D_;
  for (int i = threadIdx.x; i < D_; i += 256)
    orow[i] = f2bf(w[i] * (xr[i] - mean) * inv + b[i]);
}

// ---------- convert fp32 W[K][N] -> bf16 Wt[N][K] ----------
__global__ void k_transpose(const float* __restrict__ W, bf16* __restrict__ Wt,
                            int K, int N) {
  long idx = (long)blockIdx.x * 256 + threadIdx.x;
  int  k = (int)(idx % K);
  long n = idx / K;
  Wt[n * K + k] = f2bf(W[(long)k * N + n]);
}

// ---------- WMMA GEMM: C[M][N] = A[M][K](bf16) * Wt[N][K](bf16) + bias ----------
// Double-buffered async global->LDS pipeline.
// EPI: 0 = f32 out, 1 = gelu -> bf16 out, 2 = +residual -> f32 out
#define TBUF (128 * 40)
template<int EPI>
__global__ void __launch_bounds__(256)
k_gemm(const bf16* __restrict__ A, const bf16* __restrict__ Bt,
       const float* __restrict__ bias, const float* __restrict__ res,
       float* __restrict__ Of, bf16* __restrict__ Ob,
       int M, int N, int K) {
  __shared__ __align__(16) bf16 As[2 * TBUF];
  __shared__ __align__(16) bf16 Bs[2 * TBUF];
  const int tid  = threadIdx.x;
  const int lane = tid & 31;
  const int wave = tid >> 5;
  const int wm   = wave >> 1;          // 0..3
  const int wn   = wave & 1;           // 0..1
  const int m0   = blockIdx.y * 128;
  const int n0   = blockIdx.x * 128;
  const int lrow = tid >> 1;           // 0..127
  const int lcol = (tid & 1) * 16;     // 0 or 16
  const int half = lane >> 4;
  const int l16  = lane & 15;

  v8f acc[2][4];
  #pragma unroll
  for (int i = 0; i < 2; i++)
    #pragma unroll
    for (int j = 0; j < 4; j++) acc[i][j] = {};

  const bf16* gA = A  + (size_t)(m0 + lrow) * K + lcol;
  const bf16* gB = Bt + (size_t)(n0 + lrow) * K + lcol;
  bf16* sA = As + lrow * 40 + lcol;
  bf16* sB = Bs + lrow * 40 + lcol;

  // prefetch tile 0 into buffer 0 (4 async b128 DMAs per thread)
  async_ld_b128(gA,     sA);
  async_ld_b128(gA + 8, sA + 8);
  async_ld_b128(gB,     sB);
  async_ld_b128(gB + 8, sB + 8);

  const int nk = K >> 5;
  for (int ki = 0; ki < nk; ki++) {
    const int cur = (ki & 1) ? TBUF : 0;
    if (ki + 1 < nk) {
      const int nxt = ((ki + 1) & 1) ? TBUF : 0;
      const bf16* ga = gA + ((size_t)(ki + 1) << 5);
      const bf16* gb = gB + ((size_t)(ki + 1) << 5);
      async_ld_b128(ga,     sA + nxt);
      async_ld_b128(ga + 8, sA + nxt + 8);
      async_ld_b128(gb,     sB + nxt);
      async_ld_b128(gb + 8, sB + nxt + 8);
      // 8 outstanding; in-order completion => <=4 means current tile landed
      asm volatile("s_wait_asynccnt 0x4" ::: "memory");
    } else {
      asm volatile("s_wait_asynccnt 0x0" ::: "memory");
    }
    __syncthreads();

    FragBF af[2], bfg[4];
    #pragma unroll
    for (int mt = 0; mt < 2; mt++) {
      const bf16* p = As + cur + (wm * 32 + mt * 16 + l16) * 40;
      af[mt].u[0] = *(const u32x4*)(p + half * 8);
      af[mt].u[1] = *(const u32x4*)(p + 16 + half * 8);
    }
    #pragma unroll
    for (int nt = 0; nt < 4; nt++) {
      const bf16* p = Bs + cur + (wn * 64 + nt * 16 + l16) * 40;
      bfg[nt].u[0] = *(const u32x4*)(p + half * 8);
      bfg[nt].u[1] = *(const u32x4*)(p + 16 + half * 8);
    }
    #pragma unroll
    for (int mt = 0; mt < 2; mt++)
      #pragma unroll
      for (int nt = 0; nt < 4; nt++)
        acc[mt][nt] = __builtin_amdgcn_wmma_f32_16x16x32_bf16(
            false, af[mt].v, false, bfg[nt].v, (short)0, acc[mt][nt], false, false);

    __syncthreads();   // all waves done reading 'cur' before it is DMA-overwritten
  }

  #pragma unroll
  for (int mt = 0; mt < 2; mt++)
    #pragma unroll
    for (int nt = 0; nt < 4; nt++) {
      int gn = n0 + wn * 64 + nt * 16 + l16;
      float bv = bias[gn];
      #pragma unroll
      for (int r = 0; r < 8; r++) {
        int gm = m0 + wm * 32 + mt * 16 + half * 8 + r;
        size_t o = (size_t)gm * N + gn;
        float v = acc[mt][nt][r] + bv;
        if (EPI == 0)      Of[o] = v;
        else if (EPI == 1) Ob[o] = f2bf(gelu_f(v));
        else               Of[o] = v + res[o];
      }
    }
}

// ---------- LoRA: afterA[t][j] = sum_d h[t][d]*lA[j][d] ----------
__global__ void k_loraA(const bf16* __restrict__ h, const float* __restrict__ lA,
                        float* __restrict__ afterA) {
  int t    = blockIdx.x;
  int lane = threadIdx.x & 31;
  int j    = threadIdx.x >> 5;    // 0..7
  const bf16*  hr = h  + (size_t)t * D_;
  const float* ar = lA + (size_t)j * D_;
  float s = 0.f;
  for (int d = lane; d < D_; d += 32) s += bf2f(hr[d]) * ar[d];
  #pragma unroll
  for (int m = 16; m >= 1; m >>= 1) s += __shfl_xor(s, m, 32);
  if (lane == 0) afterA[t * 8 + j] = s;
}

// ---------- LoRA apply ----------
__global__ void k_loraApply(const float* __restrict__ afterA, const float* __restrict__ lB,
                            float* __restrict__ qkv) {
  int idx = blockIdx.x * 256 + threadIdx.x;     // over MTOK*D
  int t = idx >> 10, d = idx & 1023;
  const float* aa = afterA + (size_t)t * 8;
  float dq = 0.f, dv = 0.f;
  #pragma unroll
  for (int r = 0; r < R_; r++) {
    dq += aa[r]      * lB[(size_t)d * R_ + r];
    dv += aa[R_ + r] * lB[(size_t)(D_ + d) * R_ + r];
  }
  qkv[(size_t)t * 3 * D_ + d]          += 8.0f * dq;   // LORA_SCALE = 32/4
  qkv[(size_t)t * 3 * D_ + 2 * D_ + d] += 8.0f * dv;
}

// ---------- split qkv -> Q(scaled)/K head-major, V transposed ----------
__global__ void k_split(const float* __restrict__ qkv, bf16* __restrict__ Q,
                        bf16* __restrict__ Kc, bf16* __restrict__ Vt) {
  int idx = blockIdx.x * 256 + threadIdx.x;     // over MTOK*D
  int t = idx >> 10, d = idx & 1023;
  int b = t >> 10,  l = t & 1023;
  int h = d >> 6,   e = d & 63;
  const float* q = qkv + (size_t)t * 3 * D_;
  size_t bh = (size_t)(b * H_ + h);
  Q [(bh * L_  + l) * HD_ + e] = f2bf(q[d] * 0.125f);   // fold 1/sqrt(64)
  Kc[(bh * L_  + l) * HD_ + e] = f2bf(q[D_ + d]);
  Vt[(bh * HD_ + e) * L_  + l] = f2bf(q[2 * D_ + d]);
}

// ---------- flash attention: one wave per 16-row q-tile ----------
__global__ void __launch_bounds__(128)
k_attn(const bf16* __restrict__ Q, const bf16* __restrict__ Kc,
       const bf16* __restrict__ Vt, bf16* __restrict__ Out) {
  __shared__ __align__(16) bf16 Pl[4][16 * 40];
  const int wave = threadIdx.x >> 5;
  const int lane = threadIdx.x & 31;
  const int half = lane >> 4, l16 = lane & 15;
  const int bh = blockIdx.y;
  const int q0 = blockIdx.x * 64 + wave * 16;
  const bf16* Qh = Q  + (size_t)bh * L_ * HD_;
  const bf16* Kh = Kc + (size_t)bh * L_ * HD_;
  const bf16* Vh = Vt + (size_t)bh * HD_ * L_;
  bf16* Ps = Pl[wave];

  FragBF qa0, qa1;
  { const bf16* qr = Qh + (size_t)(q0 + l16) * HD_;
    qa0.u[0] = *(const u32x4*)(qr + half * 8);
    qa0.u[1] = *(const u32x4*)(qr + 16 + half * 8);
    qa1.u[0] = *(const u32x4*)(qr + 32 + half * 8);
    qa1.u[1] = *(const u32x4*)(qr + 48 + half * 8); }

  v8f o[4]; o[0] = {}; o[1] = {}; o[2] = {}; o[3] = {};
  float mrow[8], lrow[8];
  #pragma unroll
  for (int r = 0; r < 8; r++) { mrow[r] = -3.0e38f; lrow[r] = 0.f; }

  for (int j0 = 0; j0 < q0 + 16; j0 += 32) {
    v8f s0 = {}, s1 = {};
    FragBF kb;
    { const bf16* kr = Kh + (size_t)(j0 + l16) * HD_;
      kb.u[0] = *(const u32x4*)(kr + half * 8);
      kb.u[1] = *(const u32x4*)(kr + 16 + half * 8);
      s0 = __builtin_amdgcn_wmma_f32_16x16x32_bf16(false, qa0.v, false, kb.v, (short)0, s0, false, false);
      kb.u[0] = *(const u32x4*)(kr + 32 + half * 8);
      kb.u[1] = *(const u32x4*)(kr + 48 + half * 8);
      s0 = __builtin_amdgcn_wmma_f32_16x16x32_bf16(false, qa1.v, false, kb.v, (short)0, s0, false, false); }
    { const bf16* kr = Kh + (size_t)(j0 + 16 + l16) * HD_;
      kb.u[0] = *(const u32x4*)(kr + half * 8);
      kb.u[1] = *(const u32x4*)(kr + 16 + half * 8);
      s1 = __builtin_amdgcn_wmma_f32_16x16x32_bf16(false, qa0.v, false, kb.v, (short)0, s1, false, false);
      kb.u[0] = *(const u32x4*)(kr + 32 + half * 8);
      kb.u[1] = *(const u32x4*)(kr + 48 + half * 8);
      s1 = __builtin_amdgcn_wmma_f32_16x16x32_bf16(false, qa1.v, false, kb.v, (short)0, s1, false, false); }

    #pragma unroll
    for (int r = 0; r < 8; r++) {
      int qi = q0 + half * 8 + r;
      float v0 = (j0 + l16      <= qi) ? s0[r] : -3.0e38f;
      float v1 = (j0 + 16 + l16 <= qi) ? s1[r] : -3.0e38f;
      float mx = fmaxf(v0, v1);
      mx = fmaxf(mx, __shfl_xor(mx, 1, 32));
      mx = fmaxf(mx, __shfl_xor(mx, 2, 32));
      mx = fmaxf(mx, __shfl_xor(mx, 4, 32));
      mx = fmaxf(mx, __shfl_xor(mx, 8, 32));
      float mn = fmaxf(mrow[r], mx);
      float alpha = __expf(mrow[r] - mn);
      float p0 = __expf(v0 - mn);
      float p1 = __expf(v1 - mn);
      s0[r] = p0; s1[r] = p1;
      float rs = p0 + p1;
      rs += __shfl_xor(rs, 1, 32);
      rs += __shfl_xor(rs, 2, 32);
      rs += __shfl_xor(rs, 4, 32);
      rs += __shfl_xor(rs, 8, 32);
      lrow[r] = lrow[r] * alpha + rs;
      mrow[r] = mn;
      o[0][r] *= alpha; o[1][r] *= alpha; o[2][r] *= alpha; o[3][r] *= alpha;
    }

    // P (C-layout) -> LDS (A-layout source), bf16
    #pragma unroll
    for (int r = 0; r < 8; r++) {
      Ps[(half * 8 + r) * 40 + l16]      = f2bf(s0[r]);
      Ps[(half * 8 + r) * 40 + 16 + l16] = f2bf(s1[r]);
    }
    asm volatile("s_wait_dscnt 0x0" ::: "memory");

    FragBF pa;
    pa.u[0] = *(const u32x4*)(Ps + l16 * 40 + half * 8);
    pa.u[1] = *(const u32x4*)(Ps + l16 * 40 + 16 + half * 8);
    #pragma unroll
    for (int nt = 0; nt < 4; nt++) {
      FragBF vb;
      const bf16* vr = Vh + (size_t)(nt * 16 + l16) * L_ + j0 + half * 8;
      vb.u[0] = *(const u32x4*)(vr);
      vb.u[1] = *(const u32x4*)(vr + 16);
      o[nt] = __builtin_amdgcn_wmma_f32_16x16x32_bf16(false, pa.v, false, vb.v, (short)0, o[nt], false, false);
    }
  }

  const int b = bh >> 4, h = bh & 15;
  #pragma unroll
  for (int r = 0; r < 8; r++) {
    float inv = 1.0f / lrow[r];
    int qi = q0 + half * 8 + r;
    bf16* orow = Out + ((size_t)(b * L_ + qi)) * D_ + h * HD_;
    orow[l16]      = f2bf(o[0][r] * inv);
    orow[16 + l16] = f2bf(o[1][r] * inv);
    orow[32 + l16] = f2bf(o[2][r] * inv);
    orow[48 + l16] = f2bf(o[3][r] * inv);
  }
}

// ---------- host orchestration ----------
extern "C" void kernel_launch(void* const* d_in, const int* in_sizes, int n_in,
                              void* d_out, int out_size, void* d_ws, size_t ws_size,
                              hipStream_t stream) {
  const int*   ids   = (const int*)  d_in[0];
  const float* wte   = (const float*)d_in[1];
  const float* wpe   = (const float*)d_in[2];
  const float* ln1w  = (const float*)d_in[3];
  const float* ln1b  = (const float*)d_in[4];
  const float* attnw = (const float*)d_in[5];
  const float* attnb = (const float*)d_in[6];
  const float* loraA = (const float*)d_in[7];
  const float* loraB = (const float*)d_in[8];
  const float* projw = (const float*)d_in[9];
  const float* projb = (const float*)d_in[10];
  const float* ln2w  = (const float*)d_in[11];
  const float* ln2b  = (const float*)d_in[12];
  const float* fcw   = (const float*)d_in[13];
  const float* fcb   = (const float*)d_in[14];
  const float* fc2w  = (const float*)d_in[15];
  const float* fc2b  = (const float*)d_in[16];

  char* p = (char*)d_ws;
  auto alloc = [&](size_t bytes) -> char* {
    char* r = p; p += (bytes + 255) & ~(size_t)255; return r;
  };
  float* x      = (float*)alloc((size_t)MTOK * D_ * 4);
  bf16*  h_bf   = (bf16*) alloc((size_t)MTOK * D_ * 2);
  float* qkv    = (float*)alloc((size_t)MTOK * 3 * D_ * 4);
  float* afterA = (float*)alloc((size_t)MTOK * 8 * 4);
  bf16*  qb     = (bf16*) alloc((size_t)MTOK * D_ * 2);
  bf16*  kb     = (bf16*) alloc((size_t)MTOK * D_ * 2);
  bf16*  vtb    = (bf16*) alloc((size_t)MTOK * D_ * 2);
  bf16*  ab     = (bf16*) alloc((size_t)MTOK * D_ * 2);
  bf16*  mb     = (bf16*) alloc((size_t)MTOK * D_ * 2);
  bf16*  gb     = (bf16*) alloc((size_t)MTOK * 4 * D_ * 2);
  bf16*  wt     = (bf16*) alloc((size_t)4 * D_ * D_ * 2);   // reused per weight

  k_embed<<<MTOK, 256, 0, stream>>>(ids, wte, wpe, x);

  for (int i = 0; i < NL_; i++) {
    // ln1 -> h
    k_ln<<<MTOK, 256, 0, stream>>>(x, ln1w + i * D_, ln1b + i * D_, h_bf);
    // qkv = h @ attn_w + attn_b
    k_transpose<<<(D_ * 3 * D_) / 256, 256, 0, stream>>>(attnw + (size_t)i * D_ * 3 * D_, wt, D_, 3 * D_);
    k_gemm<0><<<dim3(3 * D_ / 128, MTOK / 128), 256, 0, stream>>>(
        h_bf, wt, attnb + (size_t)i * 3 * D_, nullptr, qkv, nullptr, MTOK, 3 * D_, D_);
    // LoRA on q and v
    k_loraA<<<MTOK, 256, 0, stream>>>(h_bf, loraA + (size_t)i * 2 * R_ * D_, afterA);
    k_loraApply<<<(MTOK * D_) / 256, 256, 0, stream>>>(afterA, loraB + (size_t)i * 2 * D_ * R_, qkv);
    // split into heads (q pre-scaled, v transposed)
    k_split<<<(MTOK * D_) / 256, 256, 0, stream>>>(qkv, qb, kb, vtb);
    // attention
    k_attn<<<dim3(L_ / 64, B_ * H_), 128, 0, stream>>>(qb, kb, vtb, ab);
    // x = x + a @ proj_w + proj_b
    k_transpose<<<(D_ * D_) / 256, 256, 0, stream>>>(projw + (size_t)i * D_ * D_, wt, D_, D_);
    k_gemm<2><<<dim3(D_ / 128, MTOK / 128), 256, 0, stream>>>(
        ab, wt, projb + (size_t)i * D_, x, x, nullptr, MTOK, D_, D_);
    // ln2 -> m
    k_ln<<<MTOK, 256, 0, stream>>>(x, ln2w + i * D_, ln2b + i * D_, mb);
    // g = gelu(m @ fc_w + fc_b)
    k_transpose<<<(D_ * 4 * D_) / 256, 256, 0, stream>>>(fcw + (size_t)i * D_ * 4 * D_, wt, D_, 4 * D_);
    k_gemm<1><<<dim3(4 * D_ / 128, MTOK / 128), 256, 0, stream>>>(
        mb, wt, fcb + (size_t)i * 4 * D_, nullptr, nullptr, gb, MTOK, 4 * D_, D_);
    // x = x + g @ fc2_w + fc2_b
    k_transpose<<<(4 * D_ * D_) / 256, 256, 0, stream>>>(fc2w + (size_t)i * 4 * D_ * D_, wt, 4 * D_, D_);
    k_gemm<2><<<dim3(D_ / 128, MTOK / 128), 256, 0, stream>>>(
        gb, wt, fc2b + (size_t)i * D_, x, x, nullptr, MTOK, D_, 4 * D_);
  }

  hipMemcpyAsync(d_out, x, (size_t)MTOK * D_ * 4, hipMemcpyDeviceToDevice, stream);
}
